// Edge_LN_L3_70574902608025
// MI455X (gfx1250) — matrix operand order repaired and checked by reading, not compiled, not run on hardware
//
#include <hip/hip_runtime.h>

#define N_NODES 50000
#define N_EDGES 800000
#define FEAT0   16
#define HDIM    128
#define NTILES  8   // HDIM / 16

typedef __attribute__((ext_vector_type(16))) __bf16 v16bf;
typedef __attribute__((ext_vector_type(8)))  __bf16 v8bf;
typedef __attribute__((ext_vector_type(8)))  float  v8f;
typedef __attribute__((ext_vector_type(4)))  unsigned int v4u;
typedef __attribute__((ext_vector_type(8)))  int v8i;
typedef __attribute__((ext_vector_type(4)))  int v4i;

// ---------------------------------------------------------------- utilities

__global__ void zero_k(float* __restrict__ p, long long n) {
  long long i = (long long)blockIdx.x * blockDim.x + threadIdx.x;
  long long stride = (long long)gridDim.x * blockDim.x;
  for (; i < n; i += stride) p[i] = 0.0f;
}

// Repack a row-major f32 weight [K x Ncols] into bf16 WMMA B-fragments.
// frag[(kc*ntiles + nt)*512 + lane*16 + e] = W[kc*32 + (lane>>4)*8 + (e&7) + (e>>3)*16][nt*16 + (lane&15)]
__global__ void convert_w_k(const float* __restrict__ src, __bf16* __restrict__ dst,
                            int K, int Ncols) {
  int tid = blockIdx.x * blockDim.x + threadIdx.x;
  if (tid >= K * Ncols) return;
  int e      = tid & 15;
  int lane   = (tid >> 4) & 31;
  int cid    = tid >> 9;
  int ntiles = Ncols >> 4;
  int kc = cid / ntiles;
  int nt = cid - kc * ntiles;
  int k = kc * 32 + ((lane >> 4) << 3) + (e & 7) + ((e >> 3) << 4);
  int n = nt * 16 + (lane & 15);
  dst[tid] = (__bf16)src[k * Ncols + n];
}

__global__ void degree_k(const long long* __restrict__ ei, float* __restrict__ deg) {
  int e = blockIdx.x * blockDim.x + threadIdx.x;
  if (e < N_EDGES) atomicAdd(&deg[(int)ei[N_EDGES + e]], 1.0f);
}

__global__ void normalize_k(float* __restrict__ buf, const float* __restrict__ deg,
                            int do_relu) {
  long long t = (long long)blockIdx.x * blockDim.x + threadIdx.x;
  if (t >= (long long)N_NODES * HDIM) return;
  int node = (int)(t >> 7);
  float d = fmaxf(deg[node], 1.0f);
  float v = buf[t] / d;
  if (do_relu) v = fmaxf(v, 0.0f);
  buf[t] = v;
}

__global__ void fc_k(const float* __restrict__ hin, const float* __restrict__ Wf,
                     const float* __restrict__ bfv, float* __restrict__ out) {
  int t = blockIdx.x * blockDim.x + threadIdx.x;
  if (t >= N_NODES * 4) return;
  int nd = t >> 2, o = t & 3;
  const float* hr = hin + (size_t)nd * HDIM;
  float acc = bfv[o];
  #pragma unroll 8
  for (int c = 0; c < HDIM; c++) acc = fmaf(hr[c], Wf[c * 4 + o], acc);
  out[t] = acc;
}

// ------------------------------------------------- TDM: flat bf16 copy global -> LDS
// One-row 2D tile descriptor: data_size=2B, tensor_dim0 = tile_dim0 = nelem,
// tensor_dim1 = tile_dim1 = 1.  Issued per-wave; tracked by TENSORcnt.
__device__ __forceinline__ void tdm_copy_to_lds(const __bf16* gsrc, __bf16* ldst,
                                                unsigned nelem) {
#if __has_builtin(__builtin_amdgcn_tensor_load_to_lds)
  unsigned long long ga = (unsigned long long)(size_t)gsrc;
  unsigned lds_off = (unsigned)(size_t)ldst;   // flat LDS aperture: addr[31:0] = LDS byte addr
  v4u g0;
  g0[0] = 1u;                                        // count=1 valid user descriptor
  g0[1] = lds_off;                                   // lds_addr (bytes)
  g0[2] = (unsigned)(ga & 0xFFFFFFFFu);              // global_addr[31:0]
  g0[3] = (unsigned)((ga >> 32) & 0x1FFFFFFu) | (2u << 30);  // addr[56:32] | type=2
  v8i g1;
  g1[0] = 0x10000;                                   // data_size=1 (2 bytes), mask=0
  g1[1] = (int)(nelem << 16);                        // tensor_dim0[15:0]  (bits 63:48)
  g1[2] = (int)(nelem >> 16) | (int)(1u << 16);      // tensor_dim0[31:16] | tensor_dim1=1
  g1[3] = (int)(nelem << 16);                        // tile_dim0 (bits 127:112)
  g1[4] = 1;                                         // tile_dim1 = 1
  g1[5] = (int)nelem;                                // tensor_dim0_stride[31:0]
  g1[6] = 0;
  g1[7] = 0;
  v4i z4 = {0, 0, 0, 0};
  v8i z8 = {0, 0, 0, 0, 0, 0, 0, 0};
  __builtin_amdgcn_tensor_load_to_lds(g0, g1, z4, z4, z8, 0);
#else
  // fallback: cooperative wave copy (32 lanes)
  const int lane = threadIdx.x & 31;
  for (unsigned i = lane * 8; i < nelem; i += 32 * 8) {
    *(v8bf*)(ldst + i) = *(const v8bf*)(gsrc + i);
  }
#endif
}

__device__ __forceinline__ void tdm_wait_all() {
#if __has_builtin(__builtin_amdgcn_s_wait_tensorcnt)
  __builtin_amdgcn_s_wait_tensorcnt(0);
#endif
}

// ------------------------------------------------- fused EdgeConv (WMMA)
// One wave per 16-edge tile; 4 waves (128 threads) per block.
// A layout (ISA 16-bit A 16x32): lane&15 = m; elems 0..7 -> K = 8*(lane>=16)+0..7,
// elems 8..15 -> +16.  C layout: vgpr v -> m = v + 8*(lane>=16), n = lane&15.
template <int FIN, bool AGG_MAX>
__global__ __launch_bounds__(128) void edgeconv_k(
    const float* __restrict__ xin,          // [N, FIN]
    const long long* __restrict__ ei,       // [2, E]  (row0 = src, row1 = dst)
    const __bf16* __restrict__ w1, const float* __restrict__ b1,   // [2*FIN x 128] frags
    const __bf16* __restrict__ w2, const float* __restrict__ b2,   // [128 x 128] frags
    float* __restrict__ out)                // [N, 128] accumulators (pre-zeroed)
{
  constexpr int K1   = 2 * FIN;
  constexpr int KC1  = K1 / 32;
  constexpr int HB   = (K1 > HDIM) ? K1 : HDIM;
  constexpr int HBE  = 4 * 16 * HB;               // h-staging elems
  constexpr int W2E  = 4 * NTILES * 512;          // 16384 elems (32 KB)
  constexpr int W1E  = (FIN == 16) ? KC1 * NTILES * 512 : 0;  // 4096 elems for layer 0
  __shared__ __bf16 smem[HBE + W2E + W1E];

  __bf16* w2lds = smem + HBE;
  __bf16* w1lds = w2lds + W2E;                    // valid only when FIN==16
  const __bf16* w1src = (FIN == 16) ? w1lds : w1; // layer-0 W1 from LDS, else global/L2

  const int wave = threadIdx.x >> 5;
  const int lane = threadIdx.x & 31;
  const long long e0 = ((long long)blockIdx.x * 4 + wave) * 16;

  // ---- kick off async weight staging (wave 0), overlapped with edge gather
  if (wave == 0) {
    tdm_copy_to_lds(w2, w2lds, W2E);
    if constexpr (FIN == 16) tdm_copy_to_lds(w1, w1lds, W1E);
  }

  // ---- stage h = [x_i , x_j - x_i] into LDS as bf16
  __bf16* hrow_w = smem + (size_t)wave * 16 * HB;
  if constexpr (FIN == 16) {
    const int m  = lane >> 1;
    const int f0 = (lane & 1) * 8;
    const long long e = e0 + m;
    const int js = (int)ei[e];
    const int jd = (int)ei[N_EDGES + e];
    const float* xip = xin + (size_t)jd * FEAT0 + f0;
    const float* xjp = xin + (size_t)js * FEAT0 + f0;
    #pragma unroll
    for (int t = 0; t < 8; t++) {
      float a = xip[t], b = xjp[t];
      hrow_w[(size_t)m * HB + f0 + t]         = (__bf16)a;
      hrow_w[(size_t)m * HB + FEAT0 + f0 + t] = (__bf16)(b - a);
    }
  } else {
    const int f0 = lane * 4;  // 4 features per lane, 32 lanes cover 128
    #pragma unroll 4
    for (int m = 0; m < 16; m++) {
      const long long e = e0 + m;
      const int js = (int)ei[e];
      const int jd = (int)ei[N_EDGES + e];
      const float4 a = *(const float4*)(xin + (size_t)jd * HDIM + f0);
      const float4 b = *(const float4*)(xin + (size_t)js * HDIM + f0);
      __bf16* hr = hrow_w + (size_t)m * HB;
      hr[f0 + 0] = (__bf16)a.x;
      hr[f0 + 1] = (__bf16)a.y;
      hr[f0 + 2] = (__bf16)a.z;
      hr[f0 + 3] = (__bf16)a.w;
      hr[HDIM + f0 + 0] = (__bf16)(b.x - a.x);
      hr[HDIM + f0 + 1] = (__bf16)(b.y - a.y);
      hr[HDIM + f0 + 2] = (__bf16)(b.z - a.z);
      hr[HDIM + f0 + 3] = (__bf16)(b.w - a.w);
    }
  }

  // destination node ids for this lane's 8 output rows (m = v + khalf)
  const int khalf = (lane >> 4) << 3;   // 0 or 8
  int nodes[8];
  #pragma unroll
  for (int v = 0; v < 8; v++) nodes[v] = (int)ei[N_EDGES + e0 + v + khalf];

  if (wave == 0) tdm_wait_all();   // TENSORcnt==0 before the block consumes weights
  __syncthreads();

  const __bf16* hrow = hrow_w + (size_t)(lane & 15) * HB;
  if (lane == 0 && FIN != 16) __builtin_prefetch(w2, 0, 0);

  v8f c[NTILES];
  #pragma unroll
  for (int nt = 0; nt < NTILES; nt++)
    #pragma unroll
    for (int q = 0; q < 8; q++) c[nt][q] = 0.0f;

  // ---- GEMM1: [16 x K1] @ [K1 x 128]
  #pragma unroll
  for (int kc = 0; kc < KC1; kc++) {
    v8bf lo = *(const v8bf*)(hrow + kc * 32 + khalf);
    v8bf hi = *(const v8bf*)(hrow + kc * 32 + 16 + khalf);
    v16bf a;
    #pragma unroll
    for (int t = 0; t < 8; t++) { a[t] = lo[t]; a[8 + t] = hi[t]; }
    #pragma unroll
    for (int nt = 0; nt < NTILES; nt++) {
      v16bf b = *(const v16bf*)(w1src + ((size_t)(kc * NTILES + nt) << 9) + (lane << 4));
      c[nt] = __builtin_amdgcn_wmma_f32_16x16x32_bf16(false, a, false, b,
                                                      (short)0, c[nt], false, false);
    }
  }

  // ---- bias + ReLU, write 16x128 activation tile back to LDS (bf16)
  #pragma unroll
  for (int nt = 0; nt < NTILES; nt++) {
    const float bv = b1[nt * 16 + (lane & 15)];
    #pragma unroll
    for (int v = 0; v < 8; v++) {
      const int m = v + khalf;
      hrow_w[(size_t)m * HB + nt * 16 + (lane & 15)] = (__bf16)fmaxf(c[nt][v] + bv, 0.0f);
    }
  }
  __syncthreads();

  // ---- GEMM2: [16 x 128] @ [128 x 128]  (weights from LDS)
  #pragma unroll
  for (int nt = 0; nt < NTILES; nt++)
    #pragma unroll
    for (int q = 0; q < 8; q++) c[nt][q] = 0.0f;

  #pragma unroll
  for (int kc = 0; kc < 4; kc++) {
    v8bf lo = *(const v8bf*)(hrow + kc * 32 + khalf);
    v8bf hi = *(const v8bf*)(hrow + kc * 32 + 16 + khalf);
    v16bf a;
    #pragma unroll
    for (int t = 0; t < 8; t++) { a[t] = lo[t]; a[8 + t] = hi[t]; }
    #pragma unroll
    for (int nt = 0; nt < NTILES; nt++) {
      v16bf b = *(const v16bf*)(w2lds + ((size_t)(kc * NTILES + nt) << 9) + (lane << 4));
      c[nt] = __builtin_amdgcn_wmma_f32_16x16x32_bf16(false, a, false, b,
                                                      (short)0, c[nt], false, false);
    }
  }

  // ---- bias + scatter aggregation at dst (L2-resident atomics)
  #pragma unroll
  for (int nt = 0; nt < NTILES; nt++) {
    const int col = nt * 16 + (lane & 15);
    const float bv = b2[col];
    #pragma unroll
    for (int v = 0; v < 8; v++) {
      float val = c[nt][v] + bv;
      float* p = out + (size_t)nodes[v] * HDIM + col;
      if constexpr (AGG_MAX) {
        // out init 0 => relu(max-agg) exactly; positive floats monotonic as ints
        if (val > 0.0f) atomicMax((int*)p, __float_as_int(val));
      } else {
        atomicAdd(p, val);
      }
    }
  }
}

// ---------------------------------------------------------------- launch

extern "C" void kernel_launch(void* const* d_in, const int* in_sizes, int n_in,
                              void* d_out, int out_size, void* d_ws, size_t ws_size,
                              hipStream_t stream) {
  (void)in_sizes; (void)n_in; (void)out_size; (void)ws_size;
  const float*     x   = (const float*)d_in[0];
  const long long* ei  = (const long long*)d_in[2];
  const float* W01 = (const float*)d_in[3];  const float* b01 = (const float*)d_in[4];
  const float* W02 = (const float*)d_in[5];  const float* b02 = (const float*)d_in[6];
  const float* W11 = (const float*)d_in[7];  const float* b11 = (const float*)d_in[8];
  const float* W12 = (const float*)d_in[9];  const float* b12 = (const float*)d_in[10];
  const float* W21 = (const float*)d_in[11]; const float* b21 = (const float*)d_in[12];
  const float* W22 = (const float*)d_in[13]; const float* b22 = (const float*)d_in[14];
  const float* Wf  = (const float*)d_in[15]; const float* bfv = (const float*)d_in[16];
  float* out = (float*)d_out;

  // workspace layout
  char* ws = (char*)d_ws;
  __bf16* wb01 = (__bf16*)ws;            // 32*128
  __bf16* wb02 = wb01 + 32 * 128;        // 128*128
  __bf16* wb11 = wb02 + 128 * 128;       // 256*128
  __bf16* wb12 = wb11 + 256 * 128;       // 128*128
  __bf16* wb21 = wb12 + 128 * 128;       // 256*128
  __bf16* wb22 = wb21 + 256 * 128;       // 128*128
  float* bufA = (float*)(ws + 262144);                 // [N, 128]
  float* bufB = bufA + (size_t)N_NODES * HDIM;         // [N, 128]
  float* deg  = bufB + (size_t)N_NODES * HDIM;         // [N]

  // zero accumulators + degree (contiguous)
  const long long ztot = (long long)N_NODES * HDIM * 2 + N_NODES;
  zero_k<<<2048, 256, 0, stream>>>(bufA, ztot);

  // repack weights -> bf16 WMMA fragments
  convert_w_k<<<(32 * 128 + 255) / 256, 256, 0, stream>>>(W01, wb01, 32, 128);
  convert_w_k<<<(128 * 128 + 255) / 256, 256, 0, stream>>>(W02, wb02, 128, 128);
  convert_w_k<<<(256 * 128 + 255) / 256, 256, 0, stream>>>(W11, wb11, 256, 128);
  convert_w_k<<<(128 * 128 + 255) / 256, 256, 0, stream>>>(W12, wb12, 128, 128);
  convert_w_k<<<(256 * 128 + 255) / 256, 256, 0, stream>>>(W21, wb21, 256, 128);
  convert_w_k<<<(128 * 128 + 255) / 256, 256, 0, stream>>>(W22, wb22, 128, 128);

  degree_k<<<(N_EDGES + 255) / 256, 256, 0, stream>>>(ei, deg);

  const int econv_grid = N_EDGES / 64;  // 4 waves x 16 edges per block

  // layer 0: max aggregation (ReLU absorbed by 0-init + positive-only atomicMax)
  edgeconv_k<16, true><<<econv_grid, 128, 0, stream>>>(x, ei, wb01, b01, wb02, b02, bufA);

  // layer 1: mean aggregation
  edgeconv_k<128, false><<<econv_grid, 128, 0, stream>>>(bufA, ei, wb11, b11, wb12, b12, bufB);
  normalize_k<<<(N_NODES * HDIM + 255) / 256, 256, 0, stream>>>(bufB, deg, 1);

  // layer 2: mean aggregation (re-zero bufA first; layer-1 read is complete)
  zero_k<<<2048, 256, 0, stream>>>(bufA, (long long)N_NODES * HDIM);
  edgeconv_k<128, false><<<econv_grid, 128, 0, stream>>>(bufB, ei, wb21, b21, wb22, b22, bufA);
  normalize_k<<<(N_NODES * HDIM + 255) / 256, 256, 0, stream>>>(bufA, deg, 0);

  // final fc: [N,128] @ [128,4] + bias
  fc_k<<<(N_NODES * 4 + 255) / 256, 256, 0, stream>>>(bufA, Wf, bfv, out);
}
